// SiSRUconv_43078521979467
// MI455X (gfx1250) — matrix operand order repaired and checked by reading, not compiled, not run on hardware
//
#include <hip/hip_runtime.h>
#include <hip/hip_bf16.h>
#include <math.h>

typedef __attribute__((ext_vector_type(16))) _Float16 v16h;
typedef __attribute__((ext_vector_type(8)))  float    v8f;

#define D_   31
#define H_   96
#define W_   96
#define HW   9216         // 96*96
#define NSP  285696       // 31*96*96
#define CIN  64
#define PD   33
#define PH   98
#define PW   98
#define PSTRIDE_D (PH*PW) // 9604
#define XT_ELEMS (PD*PH*PW*CIN)        // 20,283,648 halves (40.6 MB)
#define KSTEPS 54                      // 27 taps * 2 channel-halves of 32
#define APACK_ELEMS (16*KSTEPS*32*16)  // 442,368 halves

// B-fragment global element offset for K-step s (uniform, folds into IOFFSET)
#define BOFF_S(s) ((((s >> 1) / 9) * PSTRIDE_D + (((s >> 1) / 3) % 3) * PW + ((s >> 1) % 3)) * CIN + ((s) & 1) * 32)

// ---------------------------------------------------------------------------
// Kernel 1: x (1,64,31,96,96) f32 NCDHW -> padded f16 DHWC [33][98][98][64]
// Padding baked in so the GEMM never branches on boundaries.
// Writes are perfectly coalesced; reads hit L2 (x = 73MB < 192MB L2).
// ---------------------------------------------------------------------------
__global__ void pack_x_kernel(const float* __restrict__ x, _Float16* __restrict__ xT) {
    int idx = blockIdx.x * 256 + threadIdx.x;      // < XT_ELEMS
    int c  = idx & 63;
    int t  = idx >> 6;
    int wx = t % PW;  t /= PW;
    int hy = t % PH;
    int dz = t / PH;
    float v = 0.f;
    if (dz >= 1 && dz <= D_ && hy >= 1 && hy <= H_ && wx >= 1 && wx <= W_) {
        v = x[(((c * D_) + (dz - 1)) * H_ + (hy - 1)) * W_ + (wx - 1)];
    }
    xT[idx] = (_Float16)v;
}

// ---------------------------------------------------------------------------
// Kernel 2: weights (256,64,3,3,3) f32 -> per-lane WMMA A-fragment layout f16.
// K ordering: k = tap*64 + ci; step s covers K block [s*32, s*32+32).
// 16-bit A 16x32 layout used here (contiguous K): lane<16 holds K {0..15}; lane>=16 holds
// K {16..31}; element i maps to K = i + (lane>=16 ? 16:0).
// ---------------------------------------------------------------------------
__global__ void pack_w_kernel(const float* __restrict__ w, _Float16* __restrict__ apack) {
    int idx = blockIdx.x * 256 + threadIdx.x;      // < APACK_ELEMS
    int i  = idx & 15;
    int l  = (idx >> 4) & 31;
    int s  = (idx >> 9) % KSTEPS;
    int mt = idx / (512 * KSTEPS);
    int tap = s >> 1;
    int cb  = (s & 1) * 32;
    int kd = tap / 9, kh = (tap / 3) % 3, kw = tap % 3;
    int o  = mt * 16 + (l & 15);
    int ci = cb + i + ((l & 16) ? 16 : 0);
    apack[idx] = (_Float16)w[(((o * CIN + ci) * 3 + kd) * 3 + kh) * 3 + kw];
}

// ---------------------------------------------------------------------------
// Kernel 3: implicit-GEMM conv via v_wmma_f32_16x16x32_f16 + fused BN + act.
// Block = 256 threads = 8 waves; wave tile = 16(M) x 64(N); grid = (N/64, 2).
// Software-pipelined: step s+1 fragments are issued before step s's WMMAs so
// the compiler can use partial s_wait_loadcnt instead of a full drain.
// ---------------------------------------------------------------------------
__global__ void __launch_bounds__(256) conv_gemm_kernel(
        const _Float16* __restrict__ xT, const _Float16* __restrict__ apack,
        const float* __restrict__ gamma, const float* __restrict__ beta,
        const float* __restrict__ mean,  const float* __restrict__ var,
        _Float16* __restrict__ gates) {
    int tid  = threadIdx.x;
    int lane = tid & 31;
    int wv   = tid >> 5;
    int l15  = lane & 15;
    int hi   = (lane & 16) ? 1 : 0;
    int mbase = blockIdx.y * 128 + wv * 16;        // output-channel tile base
    int nbase = blockIdx.x * 64;                   // spatial tile base

    // Per-lane padded-tensor element offsets for the 4 N-subtiles (tap 0,0,0).
    int boff[4];
#pragma unroll
    for (int j = 0; j < 4; ++j) {
        int n  = nbase + j * 16 + l15;
        int d  = n / HW;  int rr = n - d * HW;
        int h  = rr / W_; int ww = rr - h * W_;
        boff[j] = ((d * PH + h) * PW + ww) * CIN + hi * 16;  // B K-half select
    }

    const _Float16* aptr = apack + (size_t)(mbase >> 4) * (KSTEPS * 512) + lane * 16;

    v8f zero = {0.f,0.f,0.f,0.f,0.f,0.f,0.f,0.f};
    v8f acc[4] = {zero, zero, zero, zero};

    // ---- software pipeline: preload step 0 ----
    v16h a_cur = *(const v16h*)(aptr);
    v16h b_cur[4];
#pragma unroll
    for (int j = 0; j < 4; ++j)
        b_cur[j] = *(const v16h*)(xT + boff[j] + BOFF_S(0));

#pragma unroll
    for (int s = 0; s < KSTEPS; ++s) {
        v16h a_nxt;
        v16h b_nxt[4];
        if (s + 1 < KSTEPS) {                      // issue next-step fragments
            a_nxt = *(const v16h*)(aptr + (s + 1) * 512);
            int off = BOFF_S(s + 1);               // constant after unroll
#pragma unroll
            for (int j = 0; j < 4; ++j)
                b_nxt[j] = *(const v16h*)(xT + boff[j] + off);
        }
#pragma unroll
        for (int j = 0; j < 4; ++j) {
            acc[j] = __builtin_amdgcn_wmma_f32_16x16x32_f16(
                         false, a_cur, false, b_cur[j], (short)0, acc[j], false, false);
        }
        if (s + 1 < KSTEPS) {
            a_cur = a_nxt;
#pragma unroll
            for (int j = 0; j < 4; ++j) b_cur[j] = b_nxt[j];
        }
    }

    // Epilogue: BN affine + gate nonlinearity, store gates f16 in NCDHW.
    // C/D layout: lanes 0-15 N=lane, M=r; lanes 16-31 N=lane-16, M=r+8.
    int  sel8 = hi * 8;
    int  g    = mbase >> 6;                        // gate group, wave-uniform
    bool use_tanh = (g == 0) || (g == 3);
#pragma unroll
    for (int r = 0; r < 8; ++r) {
        int m = mbase + sel8 + r;
        float sc = gamma[m] * rsqrtf(var[m] + 1e-5f);
        float sh = beta[m] - mean[m] * sc;
#pragma unroll
        for (int j = 0; j < 4; ++j) {
            float v = acc[j][r] * sc + sh;
            v = use_tanh ? tanhf(v) : 1.f / (1.f + expf(-v));
            gates[(size_t)m * NSP + (nbase + j * 16 + l15)] = (_Float16)v;
        }
    }
}

// ---------------------------------------------------------------------------
// Kernel 4: SRU recurrence over depth. One thread per (c,h,w); f32 math.
// gates rows: Wx = c, f = 64+c, r = 128+c, X = 192+c (each row NSP halves).
// ---------------------------------------------------------------------------
__global__ void recur_kernel(const _Float16* __restrict__ gates, float* __restrict__ out) {
    int t = blockIdx.x * 256 + threadIdx.x;        // < 64*HW
    int c = t / HW;
    int p = t - c * HW;
    size_t bW = (size_t)c * NSP + p;
    size_t bF = bW + (size_t) 64 * NSP;
    size_t bR = bW + (size_t)128 * NSP;
    size_t bX = bW + (size_t)192 * NSP;

    float f = (float)gates[bF];
    float r = (float)gates[bR];
    float X = (float)gates[bX];
    float C = 1.f - f;                             // t=0: Wx unused per source
    float h = r * C + (1.f - r) * X;
    size_t ob = (size_t)c * NSP + p;
    out[ob] = h;

    for (int d = 1; d < D_; ++d) {
        size_t o = (size_t)d * HW;
        float wx = (float)gates[bW + o];
        f = (float)gates[bF + o];
        r = (float)gates[bR + o];
        X = (float)gates[bX + o];
        C = f * C + (1.f - f) * wx;
        h = r * C + (1.f - r) * X;
        out[ob + o] = h;
    }
}

// ---------------------------------------------------------------------------
extern "C" void kernel_launch(void* const* d_in, const int* in_sizes, int n_in,
                              void* d_out, int out_size, void* d_ws, size_t ws_size,
                              hipStream_t stream) {
    const float* x     = (const float*)d_in[0];
    const float* w     = (const float*)d_in[1];
    const float* gamma = (const float*)d_in[2];
    const float* beta  = (const float*)d_in[3];
    const float* mean  = (const float*)d_in[4];
    const float* var   = (const float*)d_in[5];
    float* out = (float*)d_out;

    char* ws = (char*)d_ws;
    _Float16* xT    = (_Float16*)ws;
    _Float16* apack = (_Float16*)(ws + (size_t)XT_ELEMS * 2);
    _Float16* gates = (_Float16*)(ws + (size_t)XT_ELEMS * 2 + (size_t)APACK_ELEMS * 2);

    pack_x_kernel<<<XT_ELEMS / 256, 256, 0, stream>>>(x, xT);
    pack_w_kernel<<<APACK_ELEMS / 256, 256, 0, stream>>>(w, apack);
    conv_gemm_kernel<<<dim3(NSP / 64, 2), 256, 0, stream>>>(
        xT, apack, gamma, beta, mean, var, gates);
    recur_kernel<<<(64 * HW) / 256, 256, 0, stream>>>(gates, out);
}
